// EquivariantAttention_7275674599932
// MI455X (gfx1250) — matrix operand order repaired and checked by reading, not compiled, not run on hardware
//
#include <hip/hip_runtime.h>
#include <hip/hip_bf16.h>
#include <math.h>

typedef __attribute__((ext_vector_type(16))) _Float16 v16h;
typedef __attribute__((ext_vector_type(8)))  _Float16 v8h;
typedef __attribute__((ext_vector_type(8)))  float    v8f;
typedef __attribute__((ext_vector_type(4)))  float    v4f;

#define HID 128
#define NHEAD 8
#define HD 16
#define CUTOFF 5.0f
#define LN_EPS 1e-5f

// ---------------- helpers ----------------

// monotone float->uint key for atomic max
__device__ __forceinline__ unsigned fkey(float f) {
    unsigned u = __float_as_uint(f);
    return (u & 0x80000000u) ? ~u : (u | 0x80000000u);
}
__device__ __forceinline__ float funkey(unsigned k) {
    unsigned u = (k & 0x80000000u) ? (k & 0x7FFFFFFFu) : ~k;
    return __uint_as_float(u);
}

// A fragment: 16x32 f16, rows rowBase..+16 of a row-major [*, ld] matrix.
// ISA 7.12.2: lane L holds row M = L%16; half-block h = L/16 covers
// K = h*8 .. h*8+7 (halves 0..7) and K = 16+h*8 .. +7 (halves 8..15).
__device__ __forceinline__ v16h load_a_frag(const _Float16* src, int ld,
                                            int rowBase, int kBase, int lane) {
    int m = lane & 15;
    int h = lane >> 4;
    const _Float16* p = src + (size_t)(rowBase + m) * ld + kBase;
    v8h lo = *(const v8h*)(p + h * 8);
    v8h hi = *(const v8h*)(p + 16 + h * 8);
    v16h a;
#pragma unroll
    for (int i = 0; i < 8; ++i) { a[i] = lo[i]; a[8 + i] = hi[i]; }
    return a;
}

// B fragment: 32x16 f16 for D = A x B with B = W^T (row-major W [out, ld]).
// B[k][n] = W[colBase+n][kBase+k]. Lane L: n = L%16, h = L/16 covers
// K = h*8 .. h*8+7 (halves 0..7) and K = 16+h*8 .. +7 (halves 8..15).
__device__ __forceinline__ v16h load_b_frag(const _Float16* W, int ld,
                                            int colBase, int kBase, int lane) {
    int n = lane & 15;
    int h = lane >> 4;
    const _Float16* p = W + (size_t)(colBase + n) * ld + kBase;
    v8h lo = *(const v8h*)(p + h * 8);
    v8h hi = *(const v8h*)(p + 16 + h * 8);
    v16h b;
#pragma unroll
    for (int i = 0; i < 8; ++i) { b[i] = lo[i]; b[8 + i] = hi[i]; }
    return b;
}

// ---------------- kernels ----------------

// vectorized f32 -> f16 convert (8 elems/thread, b128 in/out)
__global__ void cvt_f32_to_f16_k(const float* __restrict__ src,
                                 _Float16* __restrict__ dst, int n8) {
    int i = blockIdx.x * blockDim.x + threadIdx.x;
    if (i >= n8) return;
    v4f a = *(const v4f*)(src + (size_t)i * 8);
    v4f b = *(const v4f*)(src + (size_t)i * 8 + 4);
    v8h o;
#pragma unroll
    for (int j = 0; j < 4; ++j) { o[j] = (_Float16)a[j]; o[4 + j] = (_Float16)b[j]; }
    *(v8h*)(dst + (size_t)i * 8) = o;
}

__global__ void init_max_k(unsigned* maxkey) {
    if (threadIdx.x < NHEAD) maxkey[threadIdx.x] = fkey(-INFINITY);
}

// Write one wave's 16x16 f32 tile (+bias) into a 16x128 f16 LDS tile, then
// cooperatively stream the whole block tile to global with b128 stores.
__device__ __forceinline__ void epilogue_store_f16(
    _Float16* __restrict__ tile,        // LDS, 16*HID halves
    const v8f& acc, const float* __restrict__ bias,
    _Float16* __restrict__ out, int rowBase, int colBase, int lane, int tid) {
    int n = lane & 15;
    int hi = lane >> 4;
    float bc = bias[colBase + n];
#pragma unroll
    for (int r = 0; r < 8; ++r) {
        int m = r + hi * 8;
        tile[m * HID + colBase + n] = (_Float16)(acc[r] + bc);
    }
    __syncthreads();
    // 16*128 = 2048 halves, 256 threads -> one v8h (16B) store each
    {
        int idx = tid * 8;
        int m = idx >> 7;
        int c = idx & 127;
        v8h val = *(const v8h*)(tile + idx);
        *(v8h*)(out + (size_t)(rowBase + m) * HID + c) = val;
    }
    __syncthreads();
}

// Fused QKV projection. grid = N/16, block = 256 (8 waves).
// Wave w owns the 16x16 tile at columns [16w, 16w+16) for q, k and v,
// reusing the same A fragments (12 WMMAs per wave).
__global__ void qkv_gemm_k(const _Float16* __restrict__ Xh,
                           const _Float16* __restrict__ Wqh,
                           const _Float16* __restrict__ Wkh,
                           const _Float16* __restrict__ Wvh,
                           const float* __restrict__ bq,
                           const float* __restrict__ bk,
                           const float* __restrict__ bv,
                           _Float16* __restrict__ qh,
                           _Float16* __restrict__ kh,
                           _Float16* __restrict__ vh) {
    __shared__ _Float16 tile[16 * HID];

    int rowBase = blockIdx.x * 16;
    int wave = threadIdx.x >> 5;
    int lane = threadIdx.x & 31;
    int colBase = wave * 16;
    int tid = threadIdx.x;

    // A fragments loaded once, reused for q/k/v
    v16h a[4];
#pragma unroll
    for (int i = 0; i < 4; ++i) a[i] = load_a_frag(Xh, HID, rowBase, 32 * i, lane);

    // ---- q ----
    {
        v8f acc = {};
#pragma unroll
        for (int i = 0; i < 4; ++i) {
            v16h b = load_b_frag(Wqh, HID, colBase, 32 * i, lane);
            acc = __builtin_amdgcn_wmma_f32_16x16x32_f16(
                false, a[i], false, b, (short)0, acc, false, false);
        }
        epilogue_store_f16(tile, acc, bq, qh, rowBase, colBase, lane, tid);
    }
    // ---- k ----
    {
        v8f acc = {};
#pragma unroll
        for (int i = 0; i < 4; ++i) {
            v16h b = load_b_frag(Wkh, HID, colBase, 32 * i, lane);
            acc = __builtin_amdgcn_wmma_f32_16x16x32_f16(
                false, a[i], false, b, (short)0, acc, false, false);
        }
        epilogue_store_f16(tile, acc, bk, kh, rowBase, colBase, lane, tid);
    }
    // ---- v ----
    {
        v8f acc = {};
#pragma unroll
        for (int i = 0; i < 4; ++i) {
            v16h b = load_b_frag(Wvh, HID, colBase, 32 * i, lane);
            acc = __builtin_amdgcn_wmma_f32_16x16x32_f16(
                false, a[i], false, b, (short)0, acc, false, false);
        }
        epilogue_store_f16(tile, acc, bv, vh, rowBase, colBase, lane, tid);
    }
}

// Raw attention scores + cutoff, block max-reduce, global atomicMax.
__global__ void edge_scores_k(const _Float16* __restrict__ qh,
                              const _Float16* __restrict__ kh,
                              const float* __restrict__ edge_len,
                              const long long* __restrict__ eidx,
                              const float* __restrict__ We1,
                              const float* __restrict__ be1,
                              const float* __restrict__ We2,
                              const float* __restrict__ be2,
                              float* __restrict__ attn_raw,
                              unsigned* __restrict__ maxkey, int E) {
    __shared__ unsigned smax[NHEAD];
    if (threadIdx.x < NHEAD) smax[threadIdx.x] = fkey(-INFINITY);
    __syncthreads();

    int e = blockIdx.x * blockDim.x + threadIdx.x;
    if (e < E) {
        int row = (int)eidx[e];
        int col = (int)eidx[(size_t)E + e];

        // q[row] . k[col] per head (HD=16 -> 2 chunks of 8 per head)
        float s[NHEAD];
#pragma unroll
        for (int h = 0; h < NHEAD; ++h) s[h] = 0.0f;
#pragma unroll
        for (int c = 0; c < 16; ++c) {
            v8h qa = *(const v8h*)(qh + (size_t)row * HID + c * 8);
            v8h kb = *(const v8h*)(kh + (size_t)col * HID + c * 8);
            float d = 0.0f;
#pragma unroll
            for (int i = 0; i < 8; ++i) d += (float)qa[i] * (float)kb[i];
            s[c >> 1] += d;
        }

        // edge bias MLP: silu(l*We1+be1) @ We2^T + be2
        float l = edge_len[e];
        float bias[NHEAD];
#pragma unroll
        for (int h = 0; h < NHEAD; ++h) bias[h] = be2[h];
        for (int j = 0; j < HID; ++j) {
            float t = fmaf(l, We1[j], be1[j]);
            float sig = t / (1.0f + __expf(-t));
#pragma unroll
            for (int h = 0; h < NHEAD; ++h) bias[h] = fmaf(We2[h * HID + j], sig, bias[h]);
        }

        float cut = (l < CUTOFF) ? 0.5f * (__cosf(l * (float)M_PI / CUTOFF) + 1.0f) : 0.0f;
#pragma unroll
        for (int h = 0; h < NHEAD; ++h) {
            float aa = (s[h] * 0.25f + bias[h]) * cut;   // 1/sqrt(16) = 0.25
            attn_raw[(size_t)e * NHEAD + h] = aa;
            atomicMax(&smax[h], fkey(aa));
        }
    }
    __syncthreads();
    if (threadIdx.x < NHEAD) atomicMax(&maxkey[threadIdx.x], smax[threadIdx.x]);
}

// exp(a - max), scatter into attn_sum [N,8] and out_raw [N,128].
__global__ void edge_scatter_k(const _Float16* __restrict__ vh,
                               const long long* __restrict__ eidx,
                               const float* __restrict__ attn_raw,
                               const unsigned* __restrict__ maxkey,
                               float* __restrict__ attn_sum,
                               float* __restrict__ out_raw, int E) {
    int e = blockIdx.x * blockDim.x + threadIdx.x;
    if (e >= E) return;
    int row = (int)eidx[e];
    int col = (int)eidx[(size_t)E + e];

    float w[NHEAD];
#pragma unroll
    for (int h = 0; h < NHEAD; ++h) {
        float mx = funkey(maxkey[h]);
        w[h] = __expf(attn_raw[(size_t)e * NHEAD + h] - mx);
        atomicAdd(&attn_sum[(size_t)row * NHEAD + h], w[h]);
    }
#pragma unroll
    for (int c = 0; c < 16; ++c) {
        v8h vv = *(const v8h*)(vh + (size_t)col * HID + c * 8);
        float wh = w[c >> 1];
#pragma unroll
        for (int i = 0; i < 8; ++i)
            atomicAdd(&out_raw[(size_t)row * HID + c * 8 + i], wh * (float)vv[i]);
    }
}

// out_raw / (sum + eps) -> f16
__global__ void normalize_k(const float* __restrict__ out_raw,
                            const float* __restrict__ attn_sum,
                            _Float16* __restrict__ oh, int total) {
    int i = blockIdx.x * blockDim.x + threadIdx.x;
    if (i >= total) return;
    int nrow = i >> 7;            // /128
    int d = i & 127;
    float s = attn_sum[(size_t)nrow * NHEAD + (d >> 4)] + 1e-8f;
    oh[i] = (_Float16)(out_raw[i] / s);
}

// Output projection (WMMA) + bias + residual + LayerNorm. grid = N/16, block 256.
__global__ void out_gemm_ln_k(const _Float16* __restrict__ Oh,
                              const _Float16* __restrict__ Woh,
                              const float* __restrict__ bo,
                              const float* __restrict__ x,
                              const float* __restrict__ ln_g,
                              const float* __restrict__ ln_b,
                              float* __restrict__ out) {
    __shared__ float hbuf[16][HID];
    __shared__ float mu_s[16], rs_s[16];

    int rowBase = blockIdx.x * 16;
    int wave = threadIdx.x >> 5;
    int lane = threadIdx.x & 31;
    int colBase = wave * 16;

    v8f acc = {};
#pragma unroll
    for (int kb = 0; kb < HID; kb += 32) {
        v16h a = load_a_frag(Oh, HID, rowBase, kb, lane);
        v16h b = load_b_frag(Woh, HID, colBase, kb, lane);
        acc = __builtin_amdgcn_wmma_f32_16x16x32_f16(
            false, a, false, b, (short)0, acc, false, false);
    }
    int n = lane & 15;
    int hi = lane >> 4;
    float bcol = bo[colBase + n];
#pragma unroll
    for (int r = 0; r < 8; ++r) {
        int m = r + hi * 8;
        hbuf[m][colBase + n] =
            acc[r] + bcol + x[(size_t)(rowBase + m) * HID + colBase + n];
    }
    __syncthreads();

    if (threadIdx.x < 16) {
        int m = threadIdx.x;
        float s = 0.0f;
        for (int j = 0; j < HID; ++j) s += hbuf[m][j];
        float mu = s * (1.0f / HID);
        float v = 0.0f;
        for (int j = 0; j < HID; ++j) {
            float d = hbuf[m][j] - mu;
            v += d * d;
        }
        v *= (1.0f / HID);
        mu_s[m] = mu;
        rs_s[m] = rsqrtf(v + LN_EPS);
    }
    __syncthreads();

    // 2048 f32, 256 threads -> 8 contiguous each (two b128 stores)
    {
        int idx = threadIdx.x * 8;
        int m = idx >> 7;
        int j = idx & 127;
        float* dst = out + (size_t)(rowBase + m) * HID + j;
        float mu = mu_s[m], rs = rs_s[m];
        v4f o0, o1;
#pragma unroll
        for (int i = 0; i < 4; ++i) {
            o0[i] = (hbuf[m][j + i] - mu) * rs * ln_g[j + i] + ln_b[j + i];
            o1[i] = (hbuf[m][j + 4 + i] - mu) * rs * ln_g[j + 4 + i] + ln_b[j + 4 + i];
        }
        *(v4f*)(dst) = o0;
        *(v4f*)(dst + 4) = o1;
    }
}

// ---------------- launcher ----------------

extern "C" void kernel_launch(void* const* d_in, const int* in_sizes, int n_in,
                              void* d_out, int out_size, void* d_ws, size_t ws_size,
                              hipStream_t stream) {
    const float* x        = (const float*)d_in[0];
    // d_in[1] = edge_vec (unused by reference)
    const float* edge_len = (const float*)d_in[2];
    const float* Wq = (const float*)d_in[3];  const float* bq = (const float*)d_in[4];
    const float* Wk = (const float*)d_in[5];  const float* bk = (const float*)d_in[6];
    const float* Wv = (const float*)d_in[7];  const float* bv = (const float*)d_in[8];
    const float* We1 = (const float*)d_in[9]; const float* be1 = (const float*)d_in[10];
    const float* We2 = (const float*)d_in[11];const float* be2 = (const float*)d_in[12];
    const float* Wo = (const float*)d_in[13]; const float* bo = (const float*)d_in[14];
    const float* ln_g = (const float*)d_in[15];
    const float* ln_b = (const float*)d_in[16];
    const long long* eidx = (const long long*)d_in[17];

    const int N = in_sizes[0] / HID;
    const int E = in_sizes[2];
    float* out = (float*)d_out;

    // workspace carve-out
    char* ws = (char*)d_ws;
    size_t off = 0;
    auto carve = [&](size_t bytes) -> char* {
        char* p = ws + off;
        off = (off + bytes + 255) & ~(size_t)255;
        return p;
    };
    _Float16* Xh  = (_Float16*)carve((size_t)N * HID * 2);
    _Float16* Wqh = (_Float16*)carve(HID * HID * 2);
    _Float16* Wkh = (_Float16*)carve(HID * HID * 2);
    _Float16* Wvh = (_Float16*)carve(HID * HID * 2);
    _Float16* Woh = (_Float16*)carve(HID * HID * 2);
    _Float16* qh  = (_Float16*)carve((size_t)N * HID * 2);
    _Float16* kh  = (_Float16*)carve((size_t)N * HID * 2);
    _Float16* vh  = (_Float16*)carve((size_t)N * HID * 2);
    float* attn_raw = (float*)carve((size_t)E * NHEAD * 4);
    unsigned* maxkey = (unsigned*)carve(NHEAD * 4);
    float* attn_sum = (float*)carve((size_t)N * NHEAD * 4);
    float* out_raw  = (float*)carve((size_t)N * HID * 4);
    _Float16* oh    = (_Float16*)carve((size_t)N * HID * 2);
    (void)ws_size;

    // 1) f32 -> f16 converts (vectorized, 8 elems/thread)
    {
        int n8 = N * HID / 8;
        cvt_f32_to_f16_k<<<(n8 + 255) / 256, 256, 0, stream>>>(x, Xh, n8);
        int w8 = HID * HID / 8;
        cvt_f32_to_f16_k<<<(w8 + 255) / 256, 256, 0, stream>>>(Wq, Wqh, w8);
        cvt_f32_to_f16_k<<<(w8 + 255) / 256, 256, 0, stream>>>(Wk, Wkh, w8);
        cvt_f32_to_f16_k<<<(w8 + 255) / 256, 256, 0, stream>>>(Wv, Wvh, w8);
        cvt_f32_to_f16_k<<<(w8 + 255) / 256, 256, 0, stream>>>(Wo, Woh, w8);
    }

    // 2) fused QKV projection (WMMA, A fragments reused across q/k/v)
    qkv_gemm_k<<<N / 16, 256, 0, stream>>>(Xh, Wqh, Wkh, Wvh, bq, bk, bv, qh, kh, vh);

    // 3) init accumulators
    hipMemsetAsync(attn_sum, 0, (size_t)N * NHEAD * 4, stream);
    hipMemsetAsync(out_raw, 0, (size_t)N * HID * 4, stream);
    init_max_k<<<1, 32, 0, stream>>>(maxkey);

    // 4) edge scores + global per-head max
    edge_scores_k<<<(E + 255) / 256, 256, 0, stream>>>(
        qh, kh, edge_len, eidx, We1, be1, We2, be2, attn_raw, maxkey, E);

    // 5) exp + scatter-aggregate
    edge_scatter_k<<<(E + 255) / 256, 256, 0, stream>>>(
        vh, eidx, attn_raw, maxkey, attn_sum, out_raw, E);

    // 6) normalize + convert
    {
        int n = N * HID;
        normalize_k<<<(n + 255) / 256, 256, 0, stream>>>(out_raw, attn_sum, oh, n);
    }

    // 7) output projection (WMMA) + residual + LayerNorm
    out_gemm_ln_k<<<N / 16, 256, 0, stream>>>(oh, Woh, bo, x, ln_g, ln_b, out);
}
